// ODE_RNN_4964982194676
// MI455X (gfx1250) — matrix operand order repaired and checked by reading, not compiled, not run on hardware
//
#include <hip/hip_runtime.h>

typedef __attribute__((ext_vector_type(16))) __bf16 v16bf;
typedef __attribute__((ext_vector_type(8)))  __bf16 v8bf;
typedef __attribute__((ext_vector_type(8)))  float   v8f;
typedef __attribute__((ext_vector_type(4)))  float   v4f;
typedef __bf16 bf16_t;

namespace {
constexpr int kB    = 2048;
constexpr int kT    = 200;
constexpr int kDin  = 64;
constexpr int kDlat = 64;
constexpr int kNu   = 100;   // hidden units
constexpr int kNup  = 112;   // padded to 7 N-tiles of 16
constexpr int kKc   = 128;   // concat K (= kDlat + kDin) and padded K for W2 (100->128)
constexpr int kNW   = 2;     // waves per block
constexpr int kRowsPerBlock = kNW * 16;

// LDS layout, bf16-element offsets. Weights stored transposed: Wt[n*Kpad + k].
constexpr int E_WZ1T = 0;
constexpr int E_WR1T = E_WZ1T + kNup * kKc;
constexpr int E_WH1T = E_WR1T + kNup * kKc;
constexpr int E_WF1T = E_WH1T + kNup * kKc;     // Kpad = 64
constexpr int E_WF2T = E_WF1T + kNup * kDlat;
constexpr int E_WZ2T = E_WF2T + kDlat * kKc;
constexpr int E_WR2T = E_WZ2T + kDlat * kKc;
constexpr int E_WH2T = E_WR2T + kDlat * kKc;
constexpr int E_ACT  = E_WH2T + kDlat * kKc;
constexpr int kActW  = 16 * kDlat + 16 * kKc;   // yA[16][64] + hA[16][128] per wave
constexpr int E_END  = E_ACT + kNW * kActW;
constexpr int kBiasByte = E_END * 2;
constexpr int kNBias = 4 * kNup + 4 * kDlat;
constexpr size_t kSmemBytes = (size_t)kBiasByte + (size_t)kNBias * sizeof(float);
static_assert(kB % kRowsPerBlock == 0, "batch tiling");
}

__device__ __forceinline__ v8f wmma_bf(v16bf a, v16bf b, v8f c) {
  return __builtin_amdgcn_wmma_f32_16x16x32_bf16(false, a, false, b, (short)0, c,
                                                 false, false);
}

__device__ __forceinline__ v8f zero8() {
  v8f r;
#pragma unroll
  for (int i = 0; i < 8; ++i) r[i] = 0.f;
  return r;
}

// tanh via hardware rcp: 1 - 2/(e^{2x}+1).  rcp(inf)=0 -> saturates to +/-1,
// so no clamping needed and no IEEE-division expansion.
__device__ __forceinline__ float fast_tanh(float x) {
  return __builtin_fmaf(-2.f, __builtin_amdgcn_rcpf(__expf(2.f * x) + 1.f), 1.f);
}
__device__ __forceinline__ float fast_sigm(float x) {
  return __builtin_amdgcn_rcpf(__expf(-x) + 1.f);
}

// A operand (16x32 bf16) from LDS row-major activation tile.
// lane<16: row=lane, K chunks {kt*32+0..7, kt*32+16..23}
// lane>=16: row=lane-16, K chunks {kt*32+8..15, kt*32+24..31}
__device__ __forceinline__ v16bf loadA_lds(const bf16_t* act, int stride, int r,
                                           int kt, int half) {
  const bf16_t* p = act + r * stride + kt * 32 + half * 8;
  union { v16bf v; v8bf h[2]; } u;
  u.h[0] = *(const v8bf*)(p);
  u.h[1] = *(const v8bf*)(p + 16);
  return u.v;
}

// B operand (32x16 bf16) from transposed LDS weights Wt[n*kpad + k].
// lane holds col n = lane%16, K chunks {kt*32+half*8+0..7, kt*32+16+half*8+0..7}.
__device__ __forceinline__ v16bf loadB_lds(const bf16_t* Wt, int kpad, int nt,
                                           int kt, int n, int half) {
  const bf16_t* p = Wt + (nt * 16 + n) * kpad + kt * 32 + half * 8;
  union { v16bf v; v8bf h[2]; } u;
  u.h[0] = *(const v8bf*)(p);
  u.h[1] = *(const v8bf*)(p + 16);
  return u.v;
}

// A operand built directly from a global f32 row (the input x), cvt to bf16.
__device__ __forceinline__ v16bf loadA_x(const float* xr, int kt, int half) {
  const float* p = xr + kt * 32 + half * 8;
  const v4f a = *(const v4f*)(p);
  const v4f b = *(const v4f*)(p + 4);
  const v4f c = *(const v4f*)(p + 16);
  const v4f d = *(const v4f*)(p + 20);
  v16bf r;
#pragma unroll
  for (int i = 0; i < 4; ++i) {
    r[i]      = (bf16_t)a[i];
    r[4 + i]  = (bf16_t)b[i];
    r[8 + i]  = (bf16_t)c[i];
    r[12 + i] = (bf16_t)d[i];
  }
  return r;
}

// Spill D-layout tiles (M = i + 8*half, N = nt*16 + n) into a row-major bf16
// LDS tile so they can be re-read in A layout.
__device__ __forceinline__ void storeD_lds(bf16_t* act, int stride,
                                           const v8f (&t)[4], int n, int half) {
#pragma unroll
  for (int nt = 0; nt < 4; ++nt)
#pragma unroll
    for (int i = 0; i < 8; ++i)
      act[(i + 8 * half) * stride + nt * 16 + n] = (bf16_t)t[nt][i];
}

// One gate MLP: out = act2( tanh(cA @ W1 + b1) @ W2 + b2 )
// cA = [latent A tiles a0,a1 | input A tiles xa0,xa1] (all in registers).
// Biases are folded into the activation arguments so accumulators start at 0.
template <int TANH_OUT>
__device__ __forceinline__ void gate_eval(const bf16_t* W1T, const float* b1,
                                          const bf16_t* W2T, const float* b2,
                                          v16bf a0, v16bf a1, v16bf xa0, v16bf xa1,
                                          bf16_t* hA, int n, int half,
                                          v8f (&out)[4]) {
#pragma unroll
  for (int nt = 0; nt < 7; ++nt) {
    v8f acc = zero8();
    acc = wmma_bf(a0,  loadB_lds(W1T, kKc, nt, 0, n, half), acc);
    acc = wmma_bf(a1,  loadB_lds(W1T, kKc, nt, 1, n, half), acc);
    acc = wmma_bf(xa0, loadB_lds(W1T, kKc, nt, 2, n, half), acc);
    acc = wmma_bf(xa1, loadB_lds(W1T, kKc, nt, 3, n, half), acc);
    const float bv = b1[nt * 16 + n];
#pragma unroll
    for (int i = 0; i < 8; ++i)
      hA[(i + 8 * half) * kKc + nt * 16 + n] = (bf16_t)fast_tanh(acc[i] + bv);
  }
  // hidden A tiles: load once, reuse across the 4 output N-tiles
  v16bf ah0 = loadA_lds(hA, kKc, n, 0, half);
  v16bf ah1 = loadA_lds(hA, kKc, n, 1, half);
  v16bf ah2 = loadA_lds(hA, kKc, n, 2, half);
  v16bf ah3 = loadA_lds(hA, kKc, n, 3, half);
#pragma unroll
  for (int nt = 0; nt < 4; ++nt) {
    v8f acc = zero8();
    acc = wmma_bf(ah0, loadB_lds(W2T, kKc, nt, 0, n, half), acc);
    acc = wmma_bf(ah1, loadB_lds(W2T, kKc, nt, 1, n, half), acc);
    acc = wmma_bf(ah2, loadB_lds(W2T, kKc, nt, 2, n, half), acc);
    acc = wmma_bf(ah3, loadB_lds(W2T, kKc, nt, 3, n, half), acc);
    const float bv = b2[nt * 16 + n];
#pragma unroll
    for (int i = 0; i < 8; ++i)
      out[nt][i] = TANH_OUT ? fast_tanh(acc[i] + bv) : fast_sigm(acc[i] + bv);
  }
}

__global__ __launch_bounds__(kNW * 32, 1)
void ode_rnn_wmma(const float* __restrict__ data, const float* __restrict__ ts,
                  const float* __restrict__ Wf1, const float* __restrict__ bf1,
                  const float* __restrict__ Wf2, const float* __restrict__ bf2,
                  const float* __restrict__ Wz1, const float* __restrict__ bz1,
                  const float* __restrict__ Wz2, const float* __restrict__ bz2,
                  const float* __restrict__ Wr1, const float* __restrict__ br1,
                  const float* __restrict__ Wr2, const float* __restrict__ br2,
                  const float* __restrict__ Wh1, const float* __restrict__ bh1,
                  const float* __restrict__ Wh2, const float* __restrict__ bh2,
                  float* __restrict__ out_yi, float* __restrict__ out_ys) {
  extern __shared__ char smem[];
  bf16_t* sw = (bf16_t*)smem;
  float* bias = (float*)(smem + kBiasByte);
  const int tid = threadIdx.x;

  // ---- one-time: f32 weights -> bf16 LDS, transposed to B-operand layout ----
  for (int idx = tid; idx < kNup * kKc; idx += blockDim.x) {
    const int nn = idx / kKc, kk = idx % kKc;
    const bool v = nn < kNu;
    sw[E_WZ1T + idx] = (bf16_t)(v ? Wz1[kk * kNu + nn] : 0.f);
    sw[E_WR1T + idx] = (bf16_t)(v ? Wr1[kk * kNu + nn] : 0.f);
    sw[E_WH1T + idx] = (bf16_t)(v ? Wh1[kk * kNu + nn] : 0.f);
  }
  for (int idx = tid; idx < kNup * kDlat; idx += blockDim.x) {
    const int nn = idx / kDlat, kk = idx % kDlat;
    sw[E_WF1T + idx] = (bf16_t)((nn < kNu) ? Wf1[kk * kNu + nn] : 0.f);
  }
  for (int idx = tid; idx < kDlat * kKc; idx += blockDim.x) {
    const int nn = idx / kKc, kk = idx % kKc;
    const bool v = kk < kNu;
    sw[E_WF2T + idx] = (bf16_t)(v ? Wf2[kk * kDlat + nn] : 0.f);
    sw[E_WZ2T + idx] = (bf16_t)(v ? Wz2[kk * kDlat + nn] : 0.f);
    sw[E_WR2T + idx] = (bf16_t)(v ? Wr2[kk * kDlat + nn] : 0.f);
    sw[E_WH2T + idx] = (bf16_t)(v ? Wh2[kk * kDlat + nn] : 0.f);
  }
  for (int i = tid; i < kNup; i += blockDim.x) {
    const bool v = i < kNu;
    bias[0 * kNup + i] = v ? bf1[i] : 0.f;
    bias[1 * kNup + i] = v ? bz1[i] : 0.f;
    bias[2 * kNup + i] = v ? br1[i] : 0.f;
    bias[3 * kNup + i] = v ? bh1[i] : 0.f;
  }
  for (int i = tid; i < kDlat; i += blockDim.x) {
    bias[4 * kNup + 0 * kDlat + i] = bf2[i];
    bias[4 * kNup + 1 * kDlat + i] = bz2[i];
    bias[4 * kNup + 2 * kDlat + i] = br2[i];
    bias[4 * kNup + 3 * kDlat + i] = bh2[i];
  }
  // zero the K=112..127 pad columns of every wave's hidden tile
  for (int idx = tid; idx < kNW * 16 * 16; idx += blockDim.x) {
    const int w = idx >> 8, rr = (idx >> 4) & 15, cc = idx & 15;
    sw[E_ACT + w * kActW + 16 * kDlat + rr * kKc + kNup + cc] = (bf16_t)0.f;
  }
  __syncthreads();

  const float* b_f1 = bias;
  const float* b_z1 = bias + kNup;
  const float* b_r1 = bias + 2 * kNup;
  const float* b_h1 = bias + 3 * kNup;
  const float* b_f2 = bias + 4 * kNup;
  const float* b_z2 = b_f2 + kDlat;
  const float* b_r2 = b_f2 + 2 * kDlat;
  const float* b_h2 = b_f2 + 3 * kDlat;

  const int lane = tid & 31;
  const int wv   = tid >> 5;
  const int n    = lane & 15;
  const int half = lane >> 4;
  const int rowBase = blockIdx.x * kRowsPerBlock + wv * 16;

  bf16_t* yA = sw + E_ACT + wv * kActW;   // [16][64]  latent / gated-latent A tile
  bf16_t* hA = yA + 16 * kDlat;           // [16][128] hidden A tile (padded)
  const bf16_t* Wz1T = sw + E_WZ1T;
  const bf16_t* Wr1T = sw + E_WR1T;
  const bf16_t* Wh1T = sw + E_WH1T;
  const bf16_t* Wf1T = sw + E_WF1T;
  const bf16_t* Wf2T = sw + E_WF2T;
  const bf16_t* Wz2T = sw + E_WZ2T;
  const bf16_t* Wr2T = sw + E_WR2T;
  const bf16_t* Wh2T = sw + E_WH2T;

  v8f y_d[4];
#pragma unroll
  for (int nt = 0; nt < 4; ++nt) y_d[nt] = zero8();

  const float* xrow = data + (size_t)(rowBase + n) * kT * kDin;

#pragma unroll 1
  for (int j = 1; j < kT; ++j) {
    const float dt = (j == 1) ? (ts[1] - ts[0]) : (ts[j - 2] - ts[j - 1]);
    const float* xr = xrow + (size_t)j * kDin;
    if (j + 1 < kT) __builtin_prefetch(xr + kDin, 0, 0);

    const v16bf xa0 = loadA_x(xr, 0, half);
    const v16bf xa1 = loadA_x(xr, 1, half);

    // y -> yA, pick up its two A tiles
    storeD_lds(yA, kDlat, y_d, n, half);
    const v16bf ay0 = loadA_lds(yA, kDlat, n, 0, half);
    const v16bf ay1 = loadA_lds(yA, kDlat, n, 1, half);

    // ---- ODE func: f = tanh(y Wf1 + bf1) Wf2 + bf2 ; yi_ode = y + f*dt ----
#pragma unroll
    for (int nt = 0; nt < 7; ++nt) {
      v8f acc = zero8();
      acc = wmma_bf(ay0, loadB_lds(Wf1T, kDlat, nt, 0, n, half), acc);
      acc = wmma_bf(ay1, loadB_lds(Wf1T, kDlat, nt, 1, n, half), acc);
      const float bv = b_f1[nt * 16 + n];
#pragma unroll
      for (int i = 0; i < 8; ++i)
        hA[(i + 8 * half) * kKc + nt * 16 + n] = (bf16_t)fast_tanh(acc[i] + bv);
    }
    {
      const v16bf ah0 = loadA_lds(hA, kKc, n, 0, half);
      const v16bf ah1 = loadA_lds(hA, kKc, n, 1, half);
      const v16bf ah2 = loadA_lds(hA, kKc, n, 2, half);
      const v16bf ah3 = loadA_lds(hA, kKc, n, 3, half);
#pragma unroll
      for (int nt = 0; nt < 4; ++nt) {
        v8f acc = zero8();
        acc = wmma_bf(ah0, loadB_lds(Wf2T, kKc, nt, 0, n, half), acc);
        acc = wmma_bf(ah1, loadB_lds(Wf2T, kKc, nt, 1, n, half), acc);
        acc = wmma_bf(ah2, loadB_lds(Wf2T, kKc, nt, 2, n, half), acc);
        acc = wmma_bf(ah3, loadB_lds(Wf2T, kKc, nt, 3, n, half), acc);
        const float bv = b_f2[nt * 16 + n];
#pragma unroll
        for (int i = 0; i < 8; ++i)
          y_d[nt][i] = __builtin_fmaf(acc[i] + bv, dt, y_d[nt][i]);
      }
    }
    v8f yo[4];
#pragma unroll
    for (int nt = 0; nt < 4; ++nt) yo[nt] = y_d[nt];   // yi_ode

    // yi_ode -> yA : first 64 K's of the gate concat c = [yi_ode, x]
    storeD_lds(yA, kDlat, yo, n, half);
    const v16bf ac0 = loadA_lds(yA, kDlat, n, 0, half);
    const v16bf ac1 = loadA_lds(yA, kDlat, n, 1, half);

    v8f zg[4], rg[4], hg[4];
    gate_eval<0>(Wz1T, b_z1, Wz2T, b_z2, ac0, ac1, xa0, xa1, hA, n, half, zg);
    gate_eval<0>(Wr1T, b_r1, Wr2T, b_r2, ac0, ac1, xa0, xa1, hA, n, half, rg);

    // gated latent: ch = [yi_ode * r, x] -> overwrite yA
    v8f g[4];
#pragma unroll
    for (int nt = 0; nt < 4; ++nt) g[nt] = yo[nt] * rg[nt];
    storeD_lds(yA, kDlat, g, n, half);
    const v16bf ag0 = loadA_lds(yA, kDlat, n, 0, half);
    const v16bf ag1 = loadA_lds(yA, kDlat, n, 1, half);

    gate_eval<1>(Wh1T, b_h1, Wh2T, b_h2, ag0, ag1, xa0, xa1, hA, n, half, hg);

    // y = (1-z)*h + z*yi_ode ; emit latent_ys[:, j-1, :]
#pragma unroll
    for (int nt = 0; nt < 4; ++nt) {
      v8f yn;
#pragma unroll
      for (int i = 0; i < 8; ++i)
        yn[i] = (1.f - zg[nt][i]) * hg[nt][i] + zg[nt][i] * yo[nt][i];
      y_d[nt] = yn;
#pragma unroll
      for (int i = 0; i < 8; ++i) {
        const int row = rowBase + i + 8 * half;
        out_ys[((size_t)row * (kT - 1) + (j - 1)) * kDlat + nt * 16 + n] = yn[i];
      }
    }
  }

  // final latent yi
#pragma unroll
  for (int nt = 0; nt < 4; ++nt)
#pragma unroll
    for (int i = 0; i < 8; ++i)
      out_yi[(size_t)(rowBase + i + 8 * half) * kDlat + nt * 16 + n] = y_d[nt][i];
}

extern "C" void kernel_launch(void* const* d_in, const int* in_sizes, int n_in,
                              void* d_out, int out_size, void* d_ws, size_t ws_size,
                              hipStream_t stream) {
  (void)in_sizes; (void)n_in; (void)out_size; (void)d_ws; (void)ws_size;
  const float* data = (const float*)d_in[0];
  const float* ts   = (const float*)d_in[1];
  const float* Wf1  = (const float*)d_in[2];
  const float* bf1  = (const float*)d_in[3];
  const float* Wf2  = (const float*)d_in[4];
  const float* bf2  = (const float*)d_in[5];
  const float* Wz1  = (const float*)d_in[6];
  const float* bz1  = (const float*)d_in[7];
  const float* Wz2  = (const float*)d_in[8];
  const float* bz2  = (const float*)d_in[9];
  const float* Wr1  = (const float*)d_in[10];
  const float* br1  = (const float*)d_in[11];
  const float* Wr2  = (const float*)d_in[12];
  const float* br2  = (const float*)d_in[13];
  const float* Wh1  = (const float*)d_in[14];
  const float* bh1  = (const float*)d_in[15];
  const float* Wh2  = (const float*)d_in[16];
  const float* bh2  = (const float*)d_in[17];

  float* out_yi = (float*)d_out;
  float* out_ys = out_yi + (size_t)kB * kDlat;

  dim3 grid(kB / kRowsPerBlock);
  dim3 block(kNW * 32);
  ode_rnn_wmma<<<grid, block, kSmemBytes, stream>>>(
      data, ts, Wf1, bf1, Wf2, bf2, Wz1, bz1, Wz2, bz2, Wr1, br1, Wr2, br2,
      Wh1, bh1, Wh2, bh2, out_yi, out_ys);
}
